// MultiheadWeightAttention_41807211659809
// MI455X (gfx1250) — matrix operand order repaired and checked by reading, not compiled, not run on hardware
//
#include <hip/hip_runtime.h>
#include <math.h>

// ---------------------------------------------------------------------------
// cosFormer linear attention on gfx1250 (MI455X), bf16 WMMA pipeline.
//   q = relu(query@Wq^T + bq); k = relu(key@Wk^T + bk); v = value@Wv^T + bv
//   Ks = sin_s*k, Kc = cos_s*k ; KVs = Ks^T V ; KVc = Kc^T V
//   out[l,b,m] = z * ( sin_l*(Q KVs)[l,m] + cos_l*(Q KVc)[l,m] )
//   z = 1/max(sin_l*(Q.ks_sum) + cos_l*(Q.kc_sum), eps)
// GEMMs: v_wmma_f32_16x16x32_bf16, B panels staged in LDS via async
// global->LDS copies (ASYNCcnt) when the toolchain exposes them.
// ---------------------------------------------------------------------------

typedef __attribute__((ext_vector_type(16))) __bf16 v16bf;
typedef __attribute__((ext_vector_type(8)))  float  v8f;

#define EMBED   1024
#define LQ      2048
#define SK      2048
#define NB      4
#define HALF_PI 1.57079632679489662f
#define EPS_F   1e-6f
#define INV_M   (1.0f / 2048.0f)   // 1/max(L,S)

#define BN     64                  // N-tile per block
#define BM     128                 // M-tile per block (8 waves x 16)
#define CHUNK  256                 // K elements staged per LDS chunk
#define LDP    (CHUNK + 8)         // padded LDS row stride: 528B -> b128 reads
                                   // start at bank 4*n (conflict-light)

#if __has_builtin(__builtin_amdgcn_global_load_async_to_lds_b128)
#define HAVE_ASYNC_G2L 1
#endif

// ---------------- global -> LDS staging helpers ----------------

#ifdef HAVE_ASYNC_G2L
typedef int v4i_vs __attribute__((vector_size(16)));
typedef __attribute__((address_space(1))) v4i_vs* gv4i_p;   // global v4i*
typedef __attribute__((address_space(3))) v4i_vs* lv4i_p;   // LDS v4i*
#endif

__device__ __forceinline__ void g2l_b128(const __bf16* __restrict__ g, __bf16* l) {
#ifdef HAVE_ASYNC_G2L
  __bf16* gnc = const_cast<__bf16*>(g);
  __builtin_amdgcn_global_load_async_to_lds_b128((gv4i_p)gnc, (lv4i_p)l, 0, 0);
#else
  *reinterpret_cast<uint4*>(l) = *reinterpret_cast<const uint4*>(g);
#endif
}

__device__ __forceinline__ void g2l_wait() {
#ifdef HAVE_ASYNC_G2L
#if __has_builtin(__builtin_amdgcn_s_wait_asynccnt)
  __builtin_amdgcn_s_wait_asynccnt(0);
#else
  asm volatile("s_wait_asynccnt 0x0" ::: "memory");
#endif
#endif
}

// Stage a BN x CHUNK bf16 panel (row-major, row stride ldg) into lds[BN][LDP].
// 256 threads: thread t owns row t>>2, 8 x b128 pieces at cols ((t&3)+4i)*8.
__device__ __forceinline__ void stage_panel(const __bf16* __restrict__ gbase,
                                            int ldg, __bf16* lds) {
  const int t   = threadIdx.x;
  const int row = t >> 2;
  const int c4  = t & 3;
#pragma unroll
  for (int i = 0; i < 8; ++i) {
    const int col = (c4 + 4 * i) * 8;
    g2l_b128(gbase + (size_t)row * ldg + col, lds + row * LDP + col);
  }
}

// ---------------- WMMA fragment helpers (ISA 7.12.2 layouts) ----------------

__device__ __forceinline__ v8f zero_v8f() {
  v8f z;
#pragma unroll
  for (int i = 0; i < 8; ++i) z[i] = 0.0f;
  return z;
}

// A fragment: 16x32 bf16, row = lane&15, per-lane K = grp*16 + hi*8 + j.
__device__ __forceinline__ v16bf load_a16x32(const __bf16* __restrict__ base, int ld) {
  const int lane = threadIdx.x & 31;
  const int row  = lane & 15;
  const int hi   = lane >> 4;
  const __bf16* p = base + (size_t)row * ld + hi * 8;
  union { uint4 u; __bf16 h[8]; } u0, u1;
  u0.u = *reinterpret_cast<const uint4*>(p);        // K = hi*8 + 0..7
  u1.u = *reinterpret_cast<const uint4*>(p + 16);   // K = 16 + hi*8 + 0..7
  v16bf r;
#pragma unroll
  for (int j = 0; j < 8; ++j) { r[j] = u0.h[j]; r[j + 8] = u1.h[j]; }
  return r;
}

// B fragment from LDS panel: 32x16 bf16, col n = lane&15, per-lane K = grp*16 + hi*8 + j.
// base = panel + (subtile_n0)*LDP + k_local.
__device__ __forceinline__ v16bf lds_b32x16(const __bf16* base) {
  const int lane = threadIdx.x & 31;
  const int col  = lane & 15;
  const int hi   = lane >> 4;
  const __bf16* p = base + (size_t)col * LDP + hi * 8;
  union { uint4 u; __bf16 h[8]; } u0, u1;
  u0.u = *reinterpret_cast<const uint4*>(p);        // K = hi*8 + 0..7
  u1.u = *reinterpret_cast<const uint4*>(p + 16);   // K = 16 + hi*8 + 0..7
  v16bf r;
#pragma unroll
  for (int j = 0; j < 8; ++j) { r[j] = u0.h[j]; r[j + 8] = u1.h[j]; }
  return r;
}

__device__ __forceinline__ v8f wmma_bf16(v16bf a, v16bf b, v8f c) {
  return __builtin_amdgcn_wmma_f32_16x16x32_bf16(false, a, false, b,
                                                 (short)0, c, false, false);
}

// ---------------- Pass 0: f32 -> bf16 conversion ----------------

__global__ void cvt_f32_bf16_kernel(const float* __restrict__ in,
                                    __bf16* __restrict__ out, int n) {
  int i = blockIdx.x * blockDim.x + threadIdx.x;
  if (i < n) out[i] = (__bf16)in[i];
}

// ---------------- Pass 1: projection GEMM (M=8192, N=K=1024) ----------------
// X: (L*B, E) row-major bf16 (rows m = l*NB + b). W: (N, K) row-major bf16.
// mode 0: out0[b][l][f]       = relu(v + bias)                (B, L, E)
// mode 1: out0[b][f][s]=rl*sin, out1[b][f][s]=rl*cos          (B, E, S) each
// mode 2: out0[b][f][s]       = v + bias (no relu)            (B, E, S)
__global__ void __launch_bounds__(256)
proj_gemm_kernel(const __bf16* __restrict__ X, const __bf16* __restrict__ W,
                 const float* __restrict__ bias,
                 __bf16* __restrict__ out0, __bf16* __restrict__ out1, int mode) {
  extern __shared__ __align__(16) __bf16 smem[];     // [BN][LDP]
  const int wave = threadIdx.x >> 5;                 // 0..7
  const int n0   = blockIdx.x * BN;
  const int m0   = blockIdx.y * BM + wave * 16;
  const __bf16* Arow = X + (size_t)m0 * EMBED;

  v8f acc[4];
#pragma unroll
  for (int t = 0; t < 4; ++t) acc[t] = zero_v8f();

  for (int kc = 0; kc < EMBED; kc += CHUNK) {
    stage_panel(W + (size_t)n0 * EMBED + kc, EMBED, smem);
    g2l_wait();
    __syncthreads();
    v16bf a = load_a16x32(Arow + kc, EMBED);
#pragma unroll 4
    for (int k = 0; k < CHUNK; k += 32) {
      v16bf an = a;
      if (k + 32 < CHUNK) an = load_a16x32(Arow + kc + k + 32, EMBED);
#pragma unroll
      for (int t = 0; t < 4; ++t) {
        v16bf b = lds_b32x16(smem + (size_t)(t * 16) * LDP + k);
        acc[t] = wmma_bf16(a, b, acc[t]);
      }
      a = an;
    }
    __syncthreads();
  }

  const int lane = threadIdx.x & 31;
  const int col  = lane & 15;
  const int hi   = lane >> 4;
  if (mode == 0) {
#pragma unroll
    for (int t = 0; t < 4; ++t)
#pragma unroll
      for (int r = 0; r < 8; ++r) {
        const int m = m0 + r + 8 * hi;
        const int n = n0 + t * 16 + col;
        const int l = m >> 2, b = m & 3;
        const float v = fmaxf(acc[t][r] + bias[n], 0.0f);
        out0[((size_t)b * LQ + l) * EMBED + n] = (__bf16)v;
      }
  } else if (mode == 1) {
#pragma unroll
    for (int t = 0; t < 4; ++t)
#pragma unroll
      for (int r = 0; r < 8; ++r) {
        const int m = m0 + r + 8 * hi;
        const int n = n0 + t * 16 + col;
        const int l = m >> 2, b = m & 3;
        const float v = fmaxf(acc[t][r] + bias[n], 0.0f);
        const float idx = HALF_PI * (float)(l + 1) * INV_M;
        out0[((size_t)b * EMBED + n) * SK + l] = (__bf16)(v * __sinf(idx));
        out1[((size_t)b * EMBED + n) * SK + l] = (__bf16)(v * __cosf(idx));
      }
  } else {
#pragma unroll
    for (int t = 0; t < 4; ++t)
#pragma unroll
      for (int r = 0; r < 8; ++r) {
        const int m = m0 + r + 8 * hi;
        const int n = n0 + t * 16 + col;
        const int l = m >> 2, b = m & 3;
        out0[((size_t)b * EMBED + n) * SK + l] = (__bf16)(acc[t][r] + bias[n]);
      }
  }
}

// ---------------- Pass 2a: column sums ks_sum[b][f] = sum_s KsT[b][f][s] ----

__global__ void colsum_kernel(const __bf16* __restrict__ Kt,
                              float* __restrict__ sums) {
  const __bf16* p = Kt + (size_t)blockIdx.x * SK;
  float s = 0.0f;
  for (int i = threadIdx.x; i < SK; i += 32) s += (float)p[i];
#pragma unroll
  for (int off = 16; off > 0; off >>= 1) s += __shfl_xor(s, off, 32);
  if (threadIdx.x == 0) sums[blockIdx.x] = s;
}

// ---------------- Pass 2b: z[b][l] = 1/max(sin*ds + cos*dc, eps) ------------

__global__ void zcalc_kernel(const __bf16* __restrict__ Qb,
                             const float* __restrict__ ssum,
                             const float* __restrict__ csum,
                             float* __restrict__ z) {
  const int b = blockIdx.x / LQ;
  const int l = blockIdx.x % LQ;
  const __bf16* q  = Qb + ((size_t)b * LQ + l) * EMBED;
  const float*  ss = ssum + (size_t)b * EMBED;
  const float*  cs = csum + (size_t)b * EMBED;
  float ds = 0.0f, dc = 0.0f;
  for (int i = threadIdx.x; i < EMBED; i += 32) {
    const float qv = (float)q[i];
    ds += qv * ss[i];
    dc += qv * cs[i];
  }
#pragma unroll
  for (int off = 16; off > 0; off >>= 1) {
    ds += __shfl_xor(ds, off, 32);
    dc += __shfl_xor(dc, off, 32);
  }
  if (threadIdx.x == 0) {
    const float idx = HALF_PI * (float)(l + 1) * INV_M;
    const float den = __sinf(idx) * ds + __cosf(idx) * dc;
    z[blockIdx.x] = 1.0f / fmaxf(den, EPS_F);
  }
}

// ---------------- Pass 3: KV state GEMM, per batch ---------------------------
// KVt[b][m][f] = sum_s Kt[b][f][s] * Vt[b][m][s]   (M=f, N=m, K=s=2048)
__global__ void __launch_bounds__(256)
kv_gemm_kernel(const __bf16* __restrict__ Kt, const __bf16* __restrict__ Vt,
               __bf16* __restrict__ KVt) {
  extern __shared__ __align__(16) __bf16 smem[];     // [BN][LDP]
  const int bb   = blockIdx.z;
  const int wave = threadIdx.x >> 5;
  const int n0   = blockIdx.x * BN;
  const int f0   = blockIdx.y * BM + wave * 16;
  const __bf16* A = Kt + (size_t)bb * EMBED * SK + (size_t)f0 * SK;
  const __bf16* B = Vt + (size_t)bb * EMBED * SK;

  v8f acc[4];
#pragma unroll
  for (int t = 0; t < 4; ++t) acc[t] = zero_v8f();

  for (int kc = 0; kc < SK; kc += CHUNK) {
    stage_panel(B + (size_t)n0 * SK + kc, SK, smem);
    g2l_wait();
    __syncthreads();
    v16bf a = load_a16x32(A + kc, SK);
#pragma unroll 4
    for (int k = 0; k < CHUNK; k += 32) {
      v16bf an = a;
      if (k + 32 < CHUNK) an = load_a16x32(A + kc + k + 32, SK);
#pragma unroll
      for (int t = 0; t < 4; ++t) {
        v16bf b = lds_b32x16(smem + (size_t)(t * 16) * LDP + k);
        acc[t] = wmma_bf16(a, b, acc[t]);
      }
      a = an;
    }
    __syncthreads();
  }

  const int lane = threadIdx.x & 31;
  const int col  = lane & 15;
  const int hi   = lane >> 4;
#pragma unroll
  for (int t = 0; t < 4; ++t)
#pragma unroll
    for (int r = 0; r < 8; ++r) {
      const int f = f0 + r + 8 * hi;
      const int n = n0 + t * 16 + col;
      // transposed store -> (B, m, f) so pass 4's B operand is row-contiguous
      KVt[((size_t)bb * EMBED + n) * EMBED + f] = (__bf16)acc[t][r];
    }
}

// ---------------- Pass 4: fused output dual-GEMM + normalize ----------------
// out[l, b, m] = z * ( sin_l * (Q KVs)[l,m] + cos_l * (Q KVc)[l,m] )
__global__ void __launch_bounds__(256)
out_gemm_kernel(const __bf16* __restrict__ Qb, const __bf16* __restrict__ KVst,
                const __bf16* __restrict__ KVct, const float* __restrict__ z,
                float* __restrict__ out) {
  extern __shared__ __align__(16) __bf16 smem[];     // 2 x [BN][LDP]
  __bf16* smemS = smem;
  __bf16* smemC = smem + (size_t)BN * LDP;
  const int bb   = blockIdx.z;
  const int wave = threadIdx.x >> 5;
  const int n0   = blockIdx.x * BN;
  const int l0   = blockIdx.y * BM + wave * 16;
  const __bf16* A  = Qb + ((size_t)bb * LQ + l0) * EMBED;
  const __bf16* Bs = KVst + (size_t)bb * EMBED * EMBED;
  const __bf16* Bc = KVct + (size_t)bb * EMBED * EMBED;

  v8f accS[4], accC[4];
#pragma unroll
  for (int t = 0; t < 4; ++t) { accS[t] = zero_v8f(); accC[t] = zero_v8f(); }

  for (int kc = 0; kc < EMBED; kc += CHUNK) {
    stage_panel(Bs + (size_t)n0 * EMBED + kc, EMBED, smemS);
    stage_panel(Bc + (size_t)n0 * EMBED + kc, EMBED, smemC);
    g2l_wait();
    __syncthreads();
    v16bf a = load_a16x32(A + kc, EMBED);
#pragma unroll 2
    for (int k = 0; k < CHUNK; k += 32) {
      v16bf an = a;
      if (k + 32 < CHUNK) an = load_a16x32(A + kc + k + 32, EMBED);
#pragma unroll
      for (int t = 0; t < 4; ++t) {
        v16bf bs = lds_b32x16(smemS + (size_t)(t * 16) * LDP + k);
        accS[t] = wmma_bf16(a, bs, accS[t]);
        v16bf bc = lds_b32x16(smemC + (size_t)(t * 16) * LDP + k);
        accC[t] = wmma_bf16(a, bc, accC[t]);
      }
      a = an;
    }
    __syncthreads();
  }

  const int lane = threadIdx.x & 31;
  const int col  = lane & 15;
  const int hi   = lane >> 4;
#pragma unroll
  for (int r = 0; r < 8; ++r) {
    const int l = l0 + r + 8 * hi;
    const float idx = HALF_PI * (float)(l + 1) * INV_M;
    const float sn = __sinf(idx), cn = __cosf(idx);
    const float zz = z[(size_t)bb * LQ + l];
#pragma unroll
    for (int t = 0; t < 4; ++t) {
      const int n = n0 + t * 16 + col;
      out[((size_t)l * NB + bb) * EMBED + n] =
          zz * (sn * accS[t][r] + cn * accC[t][r]);
    }
  }
}

// ---------------------------------------------------------------------------

extern "C" void kernel_launch(void* const* d_in, const int* in_sizes, int n_in,
                              void* d_out, int out_size, void* d_ws, size_t ws_size,
                              hipStream_t stream) {
  const float* query = (const float*)d_in[0];
  const float* key_  = (const float*)d_in[1];
  const float* value = (const float*)d_in[2];
  const float* Wq    = (const float*)d_in[3];
  const float* bq    = (const float*)d_in[4];
  const float* Wk    = (const float*)d_in[5];
  const float* bk    = (const float*)d_in[6];
  const float* Wv    = (const float*)d_in[7];
  const float* bv    = (const float*)d_in[8];
  float* out = (float*)d_out;

  const size_t NX = (size_t)LQ * NB * EMBED;   // 8,388,608
  const size_t NW = (size_t)EMBED * EMBED;     // 1,048,576

  char* ws = (char*)d_ws;
  size_t off = 0;
  auto alloc = [&](size_t bytes) {
    void* p = ws + off;
    off = (off + bytes + 255) & ~(size_t)255;
    return p;
  };
  __bf16* xbf  = (__bf16*)alloc(NX * 2);                       // input bf16 (reused)
  __bf16* wbf  = (__bf16*)alloc(NW * 2);                       // weight bf16 (reused)
  __bf16* Qb   = (__bf16*)alloc((size_t)NB * LQ * EMBED * 2);  // (B,L,E)
  __bf16* KsT  = (__bf16*)alloc((size_t)NB * EMBED * SK * 2);  // (B,E,S)
  __bf16* KcT  = (__bf16*)alloc((size_t)NB * EMBED * SK * 2);  // (B,E,S)
  __bf16* VT   = (__bf16*)alloc((size_t)NB * EMBED * SK * 2);  // (B,E,S)
  __bf16* KVsT = (__bf16*)alloc((size_t)NB * EMBED * EMBED * 2);
  __bf16* KVcT = (__bf16*)alloc((size_t)NB * EMBED * EMBED * 2);
  float*  ssum = (float*)alloc((size_t)NB * EMBED * 4);
  float*  csum = (float*)alloc((size_t)NB * EMBED * 4);
  float*  zb   = (float*)alloc((size_t)NB * LQ * 4);
  (void)ws_size; (void)in_sizes; (void)n_in; (void)out_size;

  const dim3 blk256(256);
  const dim3 projGrid(EMBED / BN, (LQ * NB) / BM);      // (16, 64)
  const dim3 kvGrid(EMBED / BN, EMBED / BM, NB);        // (16, 8, 4)
  const dim3 outGrid(EMBED / BN, LQ / BM, NB);          // (16, 16, 4)
  const size_t smemOne = (size_t)BN * LDP * sizeof(__bf16);   // 33,792 B
  const size_t smemTwo = 2 * smemOne;                         // 67,584 B

  // ---- Q path ----
  cvt_f32_bf16_kernel<<<(NX + 255) / 256, blk256, 0, stream>>>(query, xbf, (int)NX);
  cvt_f32_bf16_kernel<<<(NW + 255) / 256, blk256, 0, stream>>>(Wq, wbf, (int)NW);
  proj_gemm_kernel<<<projGrid, blk256, smemOne, stream>>>(xbf, wbf, bq, Qb, nullptr, 0);

  // ---- K path (emits sin- and cos-weighted transposed keys) ----
  cvt_f32_bf16_kernel<<<(NX + 255) / 256, blk256, 0, stream>>>(key_, xbf, (int)NX);
  cvt_f32_bf16_kernel<<<(NW + 255) / 256, blk256, 0, stream>>>(Wk, wbf, (int)NW);
  proj_gemm_kernel<<<projGrid, blk256, smemOne, stream>>>(xbf, wbf, bk, KsT, KcT, 1);

  // ---- V path (transposed) ----
  cvt_f32_bf16_kernel<<<(NX + 255) / 256, blk256, 0, stream>>>(value, xbf, (int)NX);
  cvt_f32_bf16_kernel<<<(NW + 255) / 256, blk256, 0, stream>>>(Wv, wbf, (int)NW);
  proj_gemm_kernel<<<projGrid, blk256, smemOne, stream>>>(xbf, wbf, bv, VT, nullptr, 2);

  // ---- normalizer pieces ----
  colsum_kernel<<<NB * EMBED, 32, 0, stream>>>(KsT, ssum);
  colsum_kernel<<<NB * EMBED, 32, 0, stream>>>(KcT, csum);
  zcalc_kernel<<<NB * LQ, 32, 0, stream>>>(Qb, ssum, csum, zb);

  // ---- KV state GEMMs ----
  kv_gemm_kernel<<<kvGrid, blk256, smemOne, stream>>>(KsT, VT, KVsT);
  kv_gemm_kernel<<<kvGrid, blk256, smemOne, stream>>>(KcT, VT, KVcT);

  // ---- fused output dual-GEMM + normalization ----
  out_gemm_kernel<<<outGrid, blk256, smemTwo, stream>>>(Qb, KVsT, KVcT, zb, out);
}
